// CellCAModel3D_11982958756105
// MI455X (gfx1250) — matrix operand order repaired and checked by reading, not compiled, not run on hardware
//
#include <hip/hip_runtime.h>
#include <hip/hip_bf16.h>

typedef _Float16 f16;
typedef _Float16 v16h __attribute__((ext_vector_type(16)));
typedef _Float16 v8h  __attribute__((ext_vector_type(8)));
typedef float    v8f  __attribute__((ext_vector_type(8)));
typedef unsigned u32x4 __attribute__((ext_vector_type(4)));
typedef int      i32x4 __attribute__((ext_vector_type(4)));
typedef int      i32x8 __attribute__((ext_vector_type(8)));

#define NC   16
#define HSZ  96
#define HSTR 96
#define DSTR (96*96)
#define CSTR (96*96*96)
#define NVOX (96*96*96)
#define ALPHA 0.1f
#define LCD  3

#define TW 16
#define TH 4
#define TD 2
#define XSW 18
#define XSH 6
#define XSD 4
#define XSN (XSW*XSH*XSD)   /* 432 halo floats per channel */

#if defined(__HIP_DEVICE_COMPILE__) && __has_builtin(__builtin_amdgcn_tensor_load_to_lds) && __has_builtin(__builtin_amdgcn_s_wait_tensorcnt)
#define USE_TDM 1
#else
#define USE_TDM 0
#endif

// ---- monotonic float<->uint encoding for atomic max over signed floats ----
__device__ __forceinline__ unsigned fenc(float f) {
  unsigned u = __float_as_uint(f);
  return (u & 0x80000000u) ? ~u : (u | 0x80000000u);
}
__device__ __forceinline__ float fdec(unsigned u) {
  return (u & 0x80000000u) ? __uint_as_float(u & 0x7fffffffu)
                           : __uint_as_float(~u);
}

__device__ __forceinline__ v8f zero8() {
  v8f z;
#pragma unroll
  for (int e = 0; e < 8; ++e) z[e] = 0.0f;
  return z;
}

// Contiguous 32B fragment load (two ds_load_b128).
__device__ __forceinline__ v16h loadFrag16(const f16* __restrict__ p) {
  const v8h* q = (const v8h*)p;
  v8h a = q[0], b = q[1];
  return __builtin_shufflevector(a, b, 0,1,2,3,4,5,6,7,8,9,10,11,12,13,14,15);
}

// A fragment (16x32 f16 MxK) from row-major src[m][64].
// ISA layout: lane l holds row M=l%16, two contiguous 8-half runs
// at K = kc*32 + (l/16)*8 and +16.  -> two ds_load_b128.
__device__ __forceinline__ v16h loadA(const f16* __restrict__ src, int lane, int kc) {
  const f16* row = src + (lane & 15) * 64 + kc * 32 + (lane >> 4) * 8;
  v8h a = *(const v8h*)row;
  v8h b = *(const v8h*)(row + 16);
  return __builtin_shufflevector(a, b, 0,1,2,3,4,5,6,7,8,9,10,11,12,13,14,15);
}

// ---------------------------------------------------------------------------
__global__ void reset_keys_kernel(unsigned* keys) {
  if (threadIdx.x < 2) keys[threadIdx.x] = 0u;
}

// 3x3x3 max-pool of channel LCD (-inf SAME padding) + global max reduction.
__global__ __launch_bounds__(256) void alive_kernel(const float* __restrict__ x,
                                                    float* __restrict__ alive,
                                                    unsigned* __restrict__ key) {
  const int idx = blockIdx.x * 256 + threadIdx.x;
  const int w = idx % HSZ, h = (idx / HSZ) % HSZ, d = idx / DSTR;
  const float* xc = x + LCD * CSTR;
  float m = -INFINITY;
#pragma unroll
  for (int dd = -1; dd <= 1; ++dd) {
    const int zd = d + dd; if ((unsigned)zd >= (unsigned)HSZ) continue;
#pragma unroll
    for (int hh = -1; hh <= 1; ++hh) {
      const int zh = h + hh; if ((unsigned)zh >= (unsigned)HSZ) continue;
#pragma unroll
      for (int ww = -1; ww <= 1; ++ww) {
        const int zw = w + ww; if ((unsigned)zw >= (unsigned)HSZ) continue;
        m = fmaxf(m, xc[zd * DSTR + zh * HSTR + zw]);
      }
    }
  }
  alive[idx] = m;

  float r = m;
#pragma unroll
  for (int off = 16; off > 0; off >>= 1) r = fmaxf(r, __shfl_xor(r, off, 32));
  __shared__ float sred[8];
  const int lane = threadIdx.x & 31, wid = threadIdx.x >> 5;
  if (lane == 0) sred[wid] = r;
  __syncthreads();
  if (threadIdx.x == 0) {
    float b = sred[0];
#pragma unroll
    for (int i = 1; i < 8; ++i) b = fmaxf(b, sred[i]);
    atomicMax(key, fenc(b));
  }
}

// Fused depthwise 3x3x3 conv (16->48) + MLP 48->64(tanh)->64(tanh)->16 (WMMA)
// + residual.  Block = 128 threads = 4 waves, tile 16x4x2 voxels.
__global__ __launch_bounds__(128) void update_kernel(
    const float* __restrict__ x,
    const float* __restrict__ wperc, const float* __restrict__ bperc,
    const float* __restrict__ w1, const float* __restrict__ b1,
    const float* __restrict__ w2, const float* __restrict__ b2,
    const float* __restrict__ w3, const float* __restrict__ b3,
    float* __restrict__ xnew)
{
  // Weights pre-swizzled into WMMA B-fragment order:
  // index = ((kc*NT + nt)*32 + lane)*16 + e ;  k = kc*32+(e>>3)*16+(lane>>4)*8+(e&7), n = nt*16+(lane&15)
  __shared__ __align__(16) f16  sW1[64 * 64];
  __shared__ __align__(16) f16  sW2[64 * 64];
  __shared__ __align__(16) f16  sW3[2 * 32 * 16];
  __shared__ __align__(16) float xs[2][XSN];       // double-buffered halo tile
  __shared__ __align__(16) f16  hbuf[4][32 * 64];  // per-wave conv out [voxel][k]
  __shared__ __align__(16) f16  stg[4][16 * 64];   // per-wave inter-layer staging
  __shared__ __align__(16) float outstg[4][16 * 16];

  const int tid  = threadIdx.x;
  const int lane = tid & 31;
  const int wv   = tid >> 5;

  for (int i = tid; i < 64 * 64; i += 128) {
    const int e = i & 15, ln = (i >> 4) & 31, nt = (i >> 9) & 3, kc = i >> 11;
    const int k = kc * 32 + (e >> 3) * 16 + (ln >> 4) * 8 + (e & 7);
    const int n = nt * 16 + (ln & 15);
    sW1[i] = (f16)((k < 48) ? w1[n * 48 + k] : 0.0f);
    sW2[i] = (f16)(w2[n * 64 + k]);
  }
  for (int i = tid; i < 2 * 32 * 16; i += 128) {
    const int e = i & 15, ln = (i >> 4) & 31, kc = (i >> 9) & 1;
    const int k = kc * 32 + (e >> 3) * 16 + (ln >> 4) * 8 + (e & 7);
    const int n = ln & 15;
    sW3[i] = (f16)(w3[n * 64 + k]);
  }

  const int wz = tid & 15;
  const int hy = (tid >> 4) & 3;
  const int dz = tid >> 6;
  const int gw = blockIdx.x * TW + wz;
  const int gh = blockIdx.y * TH + hy;
  const int gd = blockIdx.z * TD + dz;
  const int gidx = gd * DSTR + gh * HSTR + gw;

  // interior blocks: full halo is in-bounds -> TDM 3D tile load path
  const bool interior = (blockIdx.x >= 1 && blockIdx.x <= 4 &&
                         blockIdx.y >= 1 && blockIdx.y <= 22 &&
                         blockIdx.z >= 1 && blockIdx.z <= 46);
  const int lw0 = (int)blockIdx.x * TW - 1;
  const int lh0 = (int)blockIdx.y * TH - 1;
  const int ld0 = (int)blockIdx.z * TD - 1;

  // Issue halo load for channel c into buffer buf (async TDM on interior blocks).
  auto halo_load = [&](int c, int buf) {
#if USE_TDM
    if (interior) {
      if (tid < 32) {   // one wave issues the DMA (TDM ignores EXEC)
        const unsigned long long ga = (unsigned long long)(size_t)
            (x + (size_t)c * CSTR + (size_t)ld0 * DSTR + (size_t)lh0 * HSTR + lw0);
        const unsigned ldsa = (unsigned)(size_t)&xs[buf][0];  // flat LDS addr[31:0]
        u32x4 g0 = { 1u,                                   // count=1
                     ldsa,                                  // lds_addr
                     (unsigned)ga,                          // global_addr[31:0]
                     ((unsigned)(ga >> 32) & 0x01ffffffu) | 0x80000000u }; // [56:32] | type=2
        i32x8 g1 = { (int)(2u << 16),                      // data_size=4B
                     (int)(96u << 16),                     // tensor_dim0=96
                     (int)(96u << 16),                     // tensor_dim1=96 (lo16)
                     (int)(18u << 16),                     // tile_dim0=18
                     (int)(6u | (4u << 16)),               // tile_dim1=6, tile_dim2=4
                     96,                                   // tensor_dim0_stride=96
                     (int)(9216u << 16),                   // tensor_dim1_stride=9216 (lo16)
                     0 };
        i32x4 g2 = { 96, 0, 0, 0 };                        // tensor_dim2=96
        i32x4 g3 = { 0, 0, 0, 0 };
        i32x8 g4 = { 0, 0, 0, 0, 0, 0, 0, 0 };             // unused trailing words
        __builtin_amdgcn_tensor_load_to_lds(g0, g1, g2, g3, g4, 0);
      }
      return;
    }
#endif
    for (int i = tid; i < XSN; i += 128) {
      const int lw = (i % XSW) + lw0;
      const int lh = ((i / XSW) % XSH) + lh0;
      const int ld = (i / (XSW * XSH)) + ld0;
      float v = 0.0f;  // SAME zero padding for conv
      if ((unsigned)lw < (unsigned)HSZ && (unsigned)lh < (unsigned)HSZ &&
          (unsigned)ld < (unsigned)HSZ)
        v = x[c * CSTR + ld * DSTR + lh * HSTR + lw];
      xs[buf][i] = v;
    }
  };

  halo_load(0, 0);   // prologue: channel 0 in flight

  // ---- depthwise conv, software-pipelined over channels ----
  v8h hreg[6];  // 48 f16 conv outputs per voxel, register-resident
#pragma unroll
  for (int c = 0; c < NC; ++c) {
    const int cb = c & 1;
#if USE_TDM
    if (interior && tid < 32) __builtin_amdgcn_s_wait_tensorcnt(0);
#endif
    __syncthreads();   // xs[cb] ready; previous buffer's readers drained

    float nb[27];
#pragma unroll
    for (int kd = 0; kd < 3; ++kd)
#pragma unroll
      for (int kh = 0; kh < 3; ++kh)
#pragma unroll
        for (int kw = 0; kw < 3; ++kw)
          nb[kd * 9 + kh * 3 + kw] =
              xs[cb][(dz + kd) * (XSW * XSH) + (hy + kh) * XSW + (wz + kw)];

    if (c + 1 < NC) halo_load(c + 1, cb ^ 1);   // overlap DMA with FMAs below
    if (c + 1 < NC) __builtin_prefetch(&x[(c + 1) * CSTR + gidx], 0, 3);

#pragma unroll
    for (int f = 0; f < 3; ++f) {
      const float* wp = wperc + (c * 3 + f) * 27;   // uniform -> s_load
      float acc = bperc[c * 3 + f];
#pragma unroll
      for (int t = 0; t < 27; ++t) acc += wp[t] * nb[t];
      const int k = c * 3 + f;
      hreg[k >> 3][k & 7] = (f16)acc;
    }
  }
  {
    f16* hrow = &hbuf[wv][lane * 64];
#pragma unroll
    for (int j = 0; j < 6; ++j) *(v8h*)(hrow + j * 8) = hreg[j];
    v8h z{};
    *(v8h*)(hrow + 48) = z;
    *(v8h*)(hrow + 56) = z;  // K pad 48..63 = 0
  }
  __syncthreads();

  // hoisted per-lane biases
  float bb1[4], bb2[4];
#pragma unroll
  for (int nt = 0; nt < 4; ++nt) {
    bb1[nt] = b1[nt * 16 + (lane & 15)];
    bb2[nt] = b2[nt * 16 + (lane & 15)];
  }
  const float bb3 = b3[lane & 15];

  // ---- MLP via v_wmma_f32_16x16x32_f16; each wave does two 16-voxel M-tiles ----
  for (int msub = 0; msub < 2; ++msub) {
    const f16* asrc = &hbuf[wv][msub * 16 * 64];

    // layer 1: 48(->64) -> 64
    v8f acc1[4];
#pragma unroll
    for (int nt = 0; nt < 4; ++nt) acc1[nt] = zero8();
#pragma unroll
    for (int kc = 0; kc < 2; ++kc) {
      const v16h a = loadA(asrc, lane, kc);
#pragma unroll
      for (int nt = 0; nt < 4; ++nt) {
        const v16h b = loadFrag16(sW1 + ((kc * 4 + nt) * 32 + lane) * 16);
        acc1[nt] = __builtin_amdgcn_wmma_f32_16x16x32_f16(
            false, a, false, b, (short)0, acc1[nt], false, false);
      }
    }
    __syncthreads();
#pragma unroll
    for (int r = 0; r < 8; ++r) {
      const int M = r + 8 * (lane >> 4);
#pragma unroll
      for (int nt = 0; nt < 4; ++nt)
        stg[wv][M * 64 + nt * 16 + (lane & 15)] = (f16)tanhf(acc1[nt][r] + bb1[nt]);
    }
    __syncthreads();

    // layer 2: 64 -> 64
    v8f acc2[4];
#pragma unroll
    for (int nt = 0; nt < 4; ++nt) acc2[nt] = zero8();
#pragma unroll
    for (int kc = 0; kc < 2; ++kc) {
      const v16h a = loadA(&stg[wv][0], lane, kc);
#pragma unroll
      for (int nt = 0; nt < 4; ++nt) {
        const v16h b = loadFrag16(sW2 + ((kc * 4 + nt) * 32 + lane) * 16);
        acc2[nt] = __builtin_amdgcn_wmma_f32_16x16x32_f16(
            false, a, false, b, (short)0, acc2[nt], false, false);
      }
    }
    __syncthreads();
#pragma unroll
    for (int r = 0; r < 8; ++r) {
      const int M = r + 8 * (lane >> 4);
#pragma unroll
      for (int nt = 0; nt < 4; ++nt)
        stg[wv][M * 64 + nt * 16 + (lane & 15)] = (f16)tanhf(acc2[nt][r] + bb2[nt]);
    }
    __syncthreads();

    // layer 3: 64 -> 16
    v8f acc3 = zero8();
#pragma unroll
    for (int kc = 0; kc < 2; ++kc) {
      const v16h a = loadA(&stg[wv][0], lane, kc);
      const v16h b = loadFrag16(sW3 + (kc * 32 + lane) * 16);
      acc3 = __builtin_amdgcn_wmma_f32_16x16x32_f16(
          false, a, false, b, (short)0, acc3, false, false);
    }
    __syncthreads();
#pragma unroll
    for (int r = 0; r < 8; ++r) {
      const int M = r + 8 * (lane >> 4);
      outstg[wv][M * 16 + (lane & 15)] = acc3[r] + bb3;
    }
    __syncthreads();

    // residual: owner thread of each voxel in this M-tile stores 16 channels
    if (lane >= msub * 16 && lane < msub * 16 + 16) {
      const int mi = lane - msub * 16;
#pragma unroll
      for (int c = 0; c < NC; ++c)
        xnew[c * CSTR + gidx] = outstg[wv][mi * 16 + c] + x[c * CSTR + gidx];
    }
    __syncthreads();
  }
}

// life mask (pre & post) + boundary zeroing; in-place pointwise over xnew.
__global__ __launch_bounds__(256) void mask_kernel(
    const float* __restrict__ xn, const float* __restrict__ a0,
    const float* __restrict__ a1, const unsigned* __restrict__ keys,
    const int* __restrict__ inout_p, float* __restrict__ dst)
{
  const int idx = blockIdx.x * 256 + threadIdx.x;
  float t0 = ALPHA * fdec(keys[0]);
  float t1 = ALPHA * fdec(keys[1]);
  if (t0 != t0) t0 = -INFINITY;
  if (t1 != t1) t1 = -INFINITY;
  const bool life = (a0[idx] > t0) && (a1[idx] > t1);
  const int h = (idx / HSZ) % HSZ, d = idx / DSTR;
  const bool bz = (d == HSZ - 1) && (h >= inout_p[0]);
  const float s = (life && !bz) ? 1.0f : 0.0f;
#pragma unroll
  for (int c = 0; c < NC; ++c) dst[c * CSTR + idx] = s * xn[c * CSTR + idx];
}

// ---------------------------------------------------------------------------
extern "C" void kernel_launch(void* const* d_in, const int* in_sizes, int n_in,
                              void* d_out, int out_size, void* d_ws, size_t ws_size,
                              hipStream_t stream) {
  (void)in_sizes; (void)n_in; (void)out_size; (void)ws_size;
  const float* x0    = (const float*)d_in[0];
  const float* wperc = (const float*)d_in[1];
  const float* bperc = (const float*)d_in[2];
  const float* w1    = (const float*)d_in[3];
  const float* b1    = (const float*)d_in[4];
  const float* w2    = (const float*)d_in[5];
  const float* b2    = (const float*)d_in[6];
  const float* w3    = (const float*)d_in[7];
  const float* b3    = (const float*)d_in[8];
  const int*   inout_p = (const int*)d_in[12];
  float* out = (float*)d_out;

  char* ws = (char*)d_ws;
  unsigned* keys  = (unsigned*)ws;
  float* alive0 = (float*)(ws + 256);
  float* alive1 = alive0 + NVOX;
  float* xws    = alive1 + NVOX;   // ping-pong x buffer (16 * 96^3 floats)

  const dim3 ugrid(HSZ / TW, HSZ / TH, HSZ / TD);
  const float* cur = x0;
  for (int s = 0; s < 4; ++s) {
    float* nxt = (s & 1) ? out : xws;   // steps 0,2 -> ws; 1,3 -> out (final in d_out)
    reset_keys_kernel<<<1, 32, 0, stream>>>(keys);
    alive_kernel<<<NVOX / 256, 256, 0, stream>>>(cur, alive0, keys + 0);
    update_kernel<<<ugrid, 128, 0, stream>>>(cur, wperc, bperc,
                                             w1, b1, w2, b2, w3, b3, nxt);
    alive_kernel<<<NVOX / 256, 256, 0, stream>>>(nxt, alive1, keys + 1);
    mask_kernel<<<NVOX / 256, 256, 0, stream>>>(nxt, alive0, alive1, keys,
                                                inout_p, nxt);
    cur = nxt;
  }
}
